// MultiHeadSelfAttention_1219770712357
// MI455X (gfx1250) — matrix operand order repaired and checked by reading, not compiled, not run on hardware
//
#include <hip/hip_runtime.h>
#include <hip/hip_bf16.h>

// ---------------------------------------------------------------------------
// Types for CDNA5 WMMA / TDM (gfx1250, wave32)
// ---------------------------------------------------------------------------
typedef __attribute__((ext_vector_type(16))) __bf16    v16bf;
typedef __attribute__((ext_vector_type(8)))  float     v8f;
typedef __attribute__((ext_vector_type(8)))  unsigned  v8u;
typedef __attribute__((ext_vector_type(4)))  unsigned  u32x4;
typedef __attribute__((ext_vector_type(8)))  int       i32x8;
typedef __attribute__((ext_vector_type(4)))  int       i32x4;

__device__ __forceinline__ unsigned short f2bf(float f) {
    unsigned u = __builtin_bit_cast(unsigned, f);
    u += 0x7FFFu + ((u >> 16) & 1u);           // round-to-nearest-even
    return (unsigned short)(u >> 16);
}
__device__ __forceinline__ unsigned pack2(unsigned short lo, unsigned short hi) {
    return (unsigned)lo | ((unsigned)hi << 16);
}
__device__ __forceinline__ v8f wmma_bf16(v8u a, v8u b, v8f c) {
    return __builtin_amdgcn_wmma_f32_16x16x32_bf16(
        false, __builtin_bit_cast(v16bf, a),
        false, __builtin_bit_cast(v16bf, b),
        (short)0, c, false, false);
}

// ---------------------------------------------------------------------------
// TDM: 2D tile load (bf16 elements) Global -> LDS, with per-row LDS padding.
// Descriptor per CDNA5 ISA (D# group0 + group1).
// pad_interval_enc: pad after 2^enc * 8 bytes; pad_amount_enc: (enc+1) dwords.
// This toolchain exposes the 6-arg builtin:
//   (u32x4 g0, i32x8 g1, i32x4 g2, i32x4 g3, i32x8 g4, i32 cpol)
// ---------------------------------------------------------------------------
__device__ __forceinline__ void tdm_load_2d_bf16(unsigned lds_off, const void* gptr,
                                                 unsigned tile_d0, unsigned tile_d1,
                                                 unsigned long long stride0_elems,
                                                 unsigned pad_interval_enc,
                                                 unsigned pad_amount_enc) {
    unsigned long long ga = (unsigned long long)(uintptr_t)gptr;
    u32x4 g0;
    g0[0] = 1u;                                               // count=1, user D#
    g0[1] = lds_off;                                          // LDS byte address
    g0[2] = (unsigned)(ga & 0xFFFFFFFFu);                     // global_addr[31:0]
    g0[3] = (unsigned)((ga >> 32) & 0x01FFFFFFu) | (2u << 30);// [56:32] | type=2

    unsigned d0 = (1u << 16)                                  // data_size = 2B
                | (1u << 20)                                  // pad_enable
                | ((pad_interval_enc & 7u) << 22)
                | ((pad_amount_enc & 0x7Fu) << 25);
    i32x8 g1;
    g1[0] = (int)d0;
    g1[1] = (int)((tile_d0 & 0xFFFFu) << 16);                 // tensor_dim0[15:0] (== tile_d0)
    g1[2] = (int)(((tile_d0 >> 16) & 0xFFFFu) | ((tile_d1 & 0xFFFFu) << 16)); // dim0 hi | dim1 lo
    g1[3] = (int)(((tile_d1 >> 16) & 0xFFFFu) | ((tile_d0 & 0xFFFFu) << 16)); // dim1 hi | tile_dim0
    g1[4] = (int)(tile_d1 & 0xFFFFu);                         // tile_dim1 | tile_dim2=0
    g1[5] = (int)(unsigned)(stride0_elems & 0xFFFFFFFFull);   // tensor_dim0_stride lo
    g1[6] = (int)(unsigned)((stride0_elems >> 32) & 0xFFFFull); // stride hi | dim1_stride lo=0
    g1[7] = 0;
    i32x4 gz4 = {0, 0, 0, 0};
    i32x8 gz8 = {0, 0, 0, 0, 0, 0, 0, 0};
    __builtin_amdgcn_tensor_load_to_lds(g0, g1, gz4, gz4, gz8, 0);
}

// ---------------------------------------------------------------------------
// Kernel 1 & 4: C[M,N] = A[M,K] (f32) x W[N,K]^T (f32), bf16 WMMA, f32 accum
// Block tile 128x128, K staged 32 at a time through LDS (bf16, padded stride).
// 256 threads = 8 waves; wave (w&3, w>>2) owns a 32x64 tile = 2x4 WMMAs/K-step.
// ---------------------------------------------------------------------------
#define GLD 40   // padded LDS row stride (bf16) -> conflict-free, 16B-aligned rows

__global__ __launch_bounds__(256)
void gemm_xWt_bf16(const float* __restrict__ A,   // [M,K] row-major f32
                   const float* __restrict__ W,   // [N,K] row-major f32
                   float* __restrict__ C,         // [M,N] f32
                   int M, int N, int K) {
    __shared__ unsigned short sA[128 * GLD];
    __shared__ unsigned short sB[128 * GLD];

    const int tid  = threadIdx.x;
    const int lane = tid & 31;
    const int wave = tid >> 5;
    const int half = lane >> 4;
    const int l16  = lane & 15;
    const int m0   = blockIdx.y * 128;
    const int n0   = blockIdx.x * 128;
    const int wm   = (wave & 3) * 32;   // wave M offset in tile
    const int wn   = (wave >> 2) * 64;  // wave N offset in tile

    v8f acc[2][4] = {};

    for (int kt = 0; kt < K; kt += 32) {
        // ---- stage A tile (128x32 f32 -> bf16): 1024 float4, 4 per thread
#pragma unroll
        for (int i = 0; i < 4; ++i) {
            int f  = tid * 4 + i;
            int r  = f >> 3;
            int c4 = (f & 7) * 4;
            const float4 v = *(const float4*)(&A[(size_t)(m0 + r) * K + kt + c4]);
            *(unsigned*)(&sA[r * GLD + c4])     = pack2(f2bf(v.x), f2bf(v.y));
            *(unsigned*)(&sA[r * GLD + c4 + 2]) = pack2(f2bf(v.z), f2bf(v.w));
        }
        // ---- stage W tile (128x32 f32 -> bf16): 1024 float4, 4 per thread
#pragma unroll
        for (int i = 0; i < 4; ++i) {
            int f  = tid * 4 + i;
            int r  = f >> 3;
            int c4 = (f & 7) * 4;
            const float4 v = *(const float4*)(&W[(size_t)(n0 + r) * K + kt + c4]);
            *(unsigned*)(&sB[r * GLD + c4])     = pack2(f2bf(v.x), f2bf(v.y));
            *(unsigned*)(&sB[r * GLD + c4 + 2]) = pack2(f2bf(v.z), f2bf(v.w));
        }
        // prefetch next K slab while we compute this one
        if (kt + 32 < K) {
            __builtin_prefetch(&A[(size_t)(m0 + (tid >> 1)) * K + kt + 32], 0, 0);
            __builtin_prefetch(&W[(size_t)(n0 + (tid >> 1)) * K + kt + 32], 0, 0);
        }
        __syncthreads();

        // ---- fragments (ISA 7.12.2 layouts); consecutive pairs -> ds_load_b128
        v8u au[2], bu[4];
#pragma unroll
        for (int t = 0; t < 2; ++t) {
            int row = wm + t * 16 + l16;
#pragma unroll
            for (int v = 0; v < 8; ++v) {
                int ka = ((v & 4) ? 16 : 0) + half * 8 + (v & 3) * 2;
                au[t][v] = *(const unsigned*)(&sA[row * GLD + ka]);
            }
        }
#pragma unroll
        for (int j = 0; j < 4; ++j) {
            int col = wn + j * 16 + l16;
#pragma unroll
            for (int v = 0; v < 8; ++v) {
                int kb = ((v & 4) ? 16 : 0) + half * 8 + (v & 3) * 2;
                bu[j][v] = *(const unsigned*)(&sB[col * GLD + kb]);
            }
        }
#pragma unroll
        for (int i = 0; i < 2; ++i)
#pragma unroll
            for (int j = 0; j < 4; ++j)
                acc[i][j] = wmma_bf16(au[i], bu[j], acc[i][j]);
        __syncthreads();
    }

    // ---- epilogue: C-layout store (row = v + 8*half, col = l16)
#pragma unroll
    for (int i = 0; i < 2; ++i)
#pragma unroll
        for (int j = 0; j < 4; ++j)
#pragma unroll
            for (int v = 0; v < 8; ++v) {
                int row = m0 + wm + i * 16 + half * 8 + v;
                int col = n0 + wn + j * 16 + l16;
                C[(size_t)row * N + col] = acc[i][j][v];
            }
}

// ---------------------------------------------------------------------------
// Kernel 2: RoPE on Q,K + repack to bf16 head-major; V transposed to [b,h,d,l]
// ---------------------------------------------------------------------------
__global__ __launch_bounds__(256)
void rope_pack(const float* __restrict__ qkv,     // [B*L, 3*D]
               unsigned short* __restrict__ Qb,   // [B,H,L,dh] bf16
               unsigned short* __restrict__ Kb,   // [B,H,L,dh] bf16
               unsigned short* __restrict__ Vt,   // [B,H,dh,L] bf16 (transposed)
               int B, int H, int L, int dh) {
    int idx   = blockIdx.x * blockDim.x + threadIdx.x;
    int halfd = dh / 2;
    int total = B * H * L * halfd;
    if (idx >= total) return;
    int j = idx % halfd;
    int t = idx / halfd;
    int l = t % L;  t /= L;
    int h = t % H;
    int b = t / H;
    int D = H * dh;

    size_t rowOff = ((size_t)(b * L + l)) * (3 * D) + h * dh;
    float q0 = qkv[rowOff + 2 * j],         q1 = qkv[rowOff + 2 * j + 1];
    float k0 = qkv[rowOff + D + 2 * j],     k1 = qkv[rowOff + D + 2 * j + 1];
    float v0 = qkv[rowOff + 2 * D + 2 * j], v1 = qkv[rowOff + 2 * D + 2 * j + 1];

    float inv_freq = __expf(-((float)(2 * j) / (float)dh) * __logf(10000.0f));
    float ang = (float)l * inv_freq;
    float s, c;
    __sincosf(ang, &s, &c);

    size_t qkOff = (((size_t)(b * H + h)) * L + l) * dh;
    Qb[qkOff + 2 * j]     = f2bf(q0 * c - q1 * s);
    Qb[qkOff + 2 * j + 1] = f2bf(q1 * c + q0 * s);
    Kb[qkOff + 2 * j]     = f2bf(k0 * c - k1 * s);
    Kb[qkOff + 2 * j + 1] = f2bf(k1 * c + k0 * s);

    size_t vOff = ((size_t)(b * H + h)) * dh * L;
    Vt[vOff + (size_t)(2 * j) * L + l]     = f2bf(v0);
    Vt[vOff + (size_t)(2 * j + 1) * L + l] = f2bf(v1);
}

// ---------------------------------------------------------------------------
// Kernel 3: causal flash attention, bf16 WMMA, f32 accum, TDM-staged K/V.
// 128 threads = 4 waves; block owns 64 q rows (wave w -> 16 rows). Key chunks
// of 32 staged into LDS once per block via TENSOR_LOAD_TO_LDS (wave 0 issues,
// s_wait_tensorcnt, barrier). TDM row padding (+16B) keeps fragment reads
// 16B-aligned (ds_load_b128) and bank-conflict-free.
//   K tile: 32 keys x 128 dh, row 256B -> LDS stride 272B (136 bf16)
//   V tile: 128 dh x 32 keys, row  64B -> LDS stride  80B ( 40 bf16)
// ---------------------------------------------------------------------------
__global__ __launch_bounds__(128)
void attn_flash(const unsigned short* __restrict__ Qb,  // [B,H,L,dh]
                const unsigned short* __restrict__ Kb,  // [B,H,L,dh]
                const unsigned short* __restrict__ Vt,  // [B,H,dh,L]
                float* __restrict__ Y,                  // [B,L,D] f32
                int B, int H, int L, int dh) {
    const int bh   = blockIdx.y;
    const int b    = bh / H;
    const int h    = bh % H;
    const int tid  = threadIdx.x;
    const int lane = tid & 31;
    const int wave = __builtin_amdgcn_readfirstlane(tid >> 5);  // wave-uniform
    const int half = lane >> 4;
    const int l16  = lane & 15;
    const int q0b  = blockIdx.x * 64;          // block q base
    const int q0   = q0b + wave * 16;          // wave q base
    const int D    = H * dh;

    const size_t qkBase = (size_t)bh * L * dh;
    const size_t vBase  = (size_t)bh * dh * L;

    __shared__ unsigned short sK[32 * 136];          // 8704 B
    __shared__ unsigned short sV[128 * 40];          // 10240 B
    __shared__ unsigned short sP[4][16 * 32 + 16];   // per-wave P staging
    unsigned short* lp = &sP[wave][0];

    // Q fragments (A-layout), 4 dh-chunks of 32, held for the whole key loop
    v8u qa[4];
#pragma unroll
    for (int c = 0; c < 4; ++c)
#pragma unroll
        for (int v = 0; v < 8; ++v) {
            int kd = c * 32 + ((v & 4) ? 16 : 0) + half * 8 + (v & 3) * 2;
            qa[c][v] = *(const unsigned*)(&Qb[qkBase + (size_t)(q0 + l16) * dh + kd]);
        }

    v8f o[8] = {};                      // 16 x 128 output accumulator
    float rmax[8], rsum[8];
#pragma unroll
    for (int v = 0; v < 8; ++v) { rmax[v] = -1e30f; rsum[v] = 0.0f; }

    const float scale = 0.08838834764831845f;   // 1/sqrt(128)
    const int kend = q0b + 64;                  // block-uniform causal bound

    for (int kt = 0; kt < kend; kt += 32) {
        // my LDS reads from previous chunk are fully done before tiles refill
        asm volatile("s_wait_dscnt 0x0" ::: "memory");
        __syncthreads();
        if (wave == 0) {
            tdm_load_2d_bf16((unsigned)(uintptr_t)&sK[0],
                             Kb + qkBase + (size_t)kt * dh,
                             /*tile_d0=*/128, /*tile_d1=*/32,
                             /*stride0=*/(unsigned long long)dh,
                             /*pad_interval_enc=*/5, /*pad_amount_enc=*/3);
            tdm_load_2d_bf16((unsigned)(uintptr_t)&sV[0],
                             Vt + vBase + kt,
                             /*tile_d0=*/32, /*tile_d1=*/128,
                             /*stride0=*/(unsigned long long)L,
                             /*pad_interval_enc=*/3, /*pad_amount_enc=*/3);
            __builtin_amdgcn_s_wait_tensorcnt(0);
        }
        __syncthreads();

        // ---- S = Q K^T : two 16x16 tiles, 4 WMMAs each over dh=128
        v8f s[2] = {};
#pragma unroll
        for (int st = 0; st < 2; ++st) {
#pragma unroll
            for (int c = 0; c < 4; ++c) {
                v8u kb;
#pragma unroll
                for (int v = 0; v < 8; ++v) {
                    int col = c * 32 + ((v & 4) ? 16 : 0) + half * 8 + (v & 3) * 2;
                    kb[v] = *(const unsigned*)(&sK[(st * 16 + l16) * 136 + col]);
                }
                s[st] = wmma_bf16(qa[c], kb, s[st]);
            }
        }

        // ---- online softmax (per row m = half*8 + v; 16-lane reductions)
        float p0r[8], p1r[8];
#pragma unroll
        for (int v = 0; v < 8; ++v) {
            int qrow = q0 + half * 8 + v;
            float s0 = (kt + l16      <= qrow) ? s[0][v] * scale : -1e30f;
            float s1 = (kt + 16 + l16 <= qrow) ? s[1][v] * scale : -1e30f;
            float mx = fmaxf(s0, s1);
            mx = fmaxf(mx, __shfl_xor(mx, 1));
            mx = fmaxf(mx, __shfl_xor(mx, 2));
            mx = fmaxf(mx, __shfl_xor(mx, 4));
            mx = fmaxf(mx, __shfl_xor(mx, 8));
            float nmax = fmaxf(rmax[v], mx);
            float corr = __expf(rmax[v] - nmax);
            rmax[v] = nmax;
            float p0 = __expf(s0 - nmax);
            float p1 = __expf(s1 - nmax);
            float ps = p0 + p1;
            ps += __shfl_xor(ps, 1);
            ps += __shfl_xor(ps, 2);
            ps += __shfl_xor(ps, 4);
            ps += __shfl_xor(ps, 8);
            rsum[v] = rsum[v] * corr + ps;
            p0r[v] = p0; p1r[v] = p1;
#pragma unroll
            for (int t = 0; t < 8; ++t) o[t][v] *= corr;
        }

        // ---- transpose P from C-layout to A-layout via per-wave LDS
#pragma unroll
        for (int v = 0; v < 8; ++v) {
            lp[(half * 8 + v) * 32 + l16]      = f2bf(p0r[v]);
            lp[(half * 8 + v) * 32 + 16 + l16] = f2bf(p1r[v]);
        }
        asm volatile("s_wait_dscnt 0x0" ::: "memory");  // wave-local LDS RAW
        v8u pa;
#pragma unroll
        for (int v = 0; v < 8; ++v) {
            int kk = ((v & 4) ? 16 : 0) + half * 8 + (v & 3) * 2;
            pa[v] = *(const unsigned*)(&lp[l16 * 32 + kk]);
        }

        // ---- O += P V : 8 N-tiles of 16 over dh, one WMMA each
#pragma unroll
        for (int t = 0; t < 8; ++t) {
            v8u vb;
#pragma unroll
            for (int v = 0; v < 8; ++v) {
                int col = ((v & 4) ? 16 : 0) + half * 8 + (v & 3) * 2;    // local key index
                vb[v] = *(const unsigned*)(&sV[(t * 16 + l16) * 40 + col]);
            }
            o[t] = wmma_bf16(pa, vb, o[t]);
        }
    }

    // ---- finalize: divide by row sums, scatter into [B,L,D] f32
#pragma unroll
    for (int v = 0; v < 8; ++v) {
        float inv = 1.0f / rsum[v];
        int row = q0 + half * 8 + v;
#pragma unroll
        for (int t = 0; t < 8; ++t)
            Y[((size_t)(b * L + row)) * D + h * dh + t * 16 + l16] = o[t][v] * inv;
    }
}

// ---------------------------------------------------------------------------
// Host launcher
// ---------------------------------------------------------------------------
extern "C" void kernel_launch(void* const* d_in, const int* in_sizes, int n_in,
                              void* d_out, int out_size, void* d_ws, size_t ws_size,
                              hipStream_t stream) {
    (void)in_sizes; (void)n_in; (void)out_size; (void)ws_size;
    const float* x    = (const float*)d_in[0];   // [B,L,D]
    const float* Wqkv = (const float*)d_in[1];   // [3D,D]
    const float* Wo   = (const float*)d_in[2];   // [D,D]
    float* out = (float*)d_out;                  // [B,L,D]

    const int B = 2, L = 2048, D = 2048, H = 16, dh = 128;
    const int M = B * L;

    // workspace carve-out
    char* ws = (char*)d_ws;
    float* qkv = (float*)ws;                                   // [M, 3D] f32
    size_t off = (size_t)M * 3 * D * sizeof(float);
    unsigned short* Qb = (unsigned short*)(ws + off); off += (size_t)M * D * 2;
    unsigned short* Kb = (unsigned short*)(ws + off); off += (size_t)M * D * 2;
    unsigned short* Vt = (unsigned short*)(ws + off); off += (size_t)M * D * 2;
    float* y = (float*)(ws + off);                             // [M, D] f32

    // 1) qkv = x @ W_qkv^T
    dim3 g1((3 * D) / 128, M / 128);
    gemm_xWt_bf16<<<g1, 256, 0, stream>>>(x, Wqkv, qkv, M, 3 * D, D);

    // 2) RoPE + repack to bf16 (Q,K) and transposed bf16 V
    int total = B * H * L * (dh / 2);
    rope_pack<<<(total + 255) / 256, 256, 0, stream>>>(qkv, Qb, Kb, Vt, B, H, L, dh);

    // 3) causal flash attention (TDM-staged K/V tiles)
    dim3 g3(L / 64, B * H);
    attn_flash<<<g3, 128, 0, stream>>>(Qb, Kb, Vt, y, B, H, L, dh);

    // 4) out = y @ W_o^T
    dim3 g4(D / 128, M / 128);
    gemm_xWt_bf16<<<g4, 256, 0, stream>>>(y, Wo, out, M, D, D);
}
